// ResidualQuantizationLayer_5068061409707
// MI455X (gfx1250) — matrix operand order repaired and checked by reading, not compiled, not run on hardware
//
#include <hip/hip_runtime.h>

typedef unsigned short u16;
typedef __attribute__((ext_vector_type(16))) __bf16 v16bf;
typedef __attribute__((ext_vector_type(8)))  float  v8f;

#define BB 8192
#define DD 768
#define KK 8192
#define LL 4
#define NT 4        // code tiles (of 16) per wave pass -> 64 codes / n0 iteration

union Frag { v16bf bf; uint4 q[2]; };

__device__ inline u16 f2bf(float f) {
    unsigned int u = __float_as_uint(f);
    unsigned int r = u + 0x7FFFu + ((u >> 16) & 1u);   // round-to-nearest-even
    return (u16)(r >> 16);
}

// ---------------------------------------------------------------------------
// Level 0 init: residual = x, zero quantized_x accumulator + scalar outputs.
// ---------------------------------------------------------------------------
__global__ void rvq_init_kernel(const float* __restrict__ x, float* __restrict__ Rf32,
                                float* __restrict__ qx, float* __restrict__ smalls,
                                float* __restrict__ loss) {
    int t = blockIdx.x * blockDim.x + threadIdx.x;
    if (t < BB * DD) {
        Rf32[t] = x[t];
        qx[t]   = 0.0f;
    }
    if (t == 0) { smalls[0] = 0.0f; loss[0] = 0.0f; }
}

// ---------------------------------------------------------------------------
// Per level: 32x32 LDS-tiled transpose of codebook [D,K] f32 ->
//   EbfT [K,D] bf16 (WMMA B-matrix)  +  ET32 [K,D] f32 (coalesced gather rows)
// Coalesced reads (along K) and coalesced writes (along D).
// ---------------------------------------------------------------------------
__global__ void rvq_transpose_kernel(const float* __restrict__ E, u16* __restrict__ EbfT,
                                     float* __restrict__ ET32) {
    __shared__ float tile[32][33];                 // pad to kill bank conflicts
    const int k0 = blockIdx.x * 32;
    const int d0 = blockIdx.y * 32;
    const int tx = threadIdx.x;                    // 0..31
    const int ty = threadIdx.y;                    // 0..7
#pragma unroll
    for (int j = 0; j < 4; ++j) {
        int dl = ty + 8 * j;
        tile[dl][tx] = E[(size_t)(d0 + dl) * KK + k0 + tx];
    }
    __syncthreads();
#pragma unroll
    for (int j = 0; j < 4; ++j) {
        int kl = ty + 8 * j;
        float v = tile[tx][kl];                    // (d=tx, k=kl)
        size_t o = (size_t)(k0 + kl) * DD + d0 + tx;
        ET32[o] = v;
        EbfT[o] = f2bf(v);
    }
}

// ---------------------------------------------------------------------------
// Per level: ||E_k||^2 + small-cluster count.
// ---------------------------------------------------------------------------
__global__ void rvq_norms_kernel(const float* __restrict__ E, const float* __restrict__ cs,
                                 float* __restrict__ norms, float* __restrict__ smalls) {
    int k = blockIdx.x * blockDim.x + threadIdx.x;
    float cnt = 0.0f;
    if (k < KK) {
        float s = 0.0f;
        for (int d = 0; d < DD; ++d) {
            float v = E[(size_t)d * KK + k];       // coalesced across threads
            s += v * v;
        }
        norms[k] = s;
        cnt = (cs[k] < 1.0f) ? 1.0f : 0.0f;
    }
    for (int off = 16; off > 0; off >>= 1) cnt += __shfl_down(cnt, off, 32);
    __shared__ float wsum[8];
    int lane = threadIdx.x & 31, wid = threadIdx.x >> 5;
    if (lane == 0) wsum[wid] = cnt;
    __syncthreads();
    if (threadIdx.x == 0) {
        float s = 0.0f;
        for (int w = 0; w < (int)(blockDim.x >> 5); ++w) s += wsum[w];
        atomicAdd(smalls, s);
    }
}

// ---------------------------------------------------------------------------
// Per level: residual f32 -> bf16.
// ---------------------------------------------------------------------------
__global__ void rvq_tobf16_kernel(const float* __restrict__ Rf32, u16* __restrict__ Rbf) {
    int t = blockIdx.x * blockDim.x + threadIdx.x;
    if (t < BB * DD) Rbf[t] = f2bf(Rf32[t]);
}

// ---------------------------------------------------------------------------
// Main: fused bf16-WMMA GEMM + per-row argmin over all K codes.
// Block = 128 threads (4 waves, wave32), 64 residual rows staged in 96KB LDS.
// Each wave owns a 16-row stripe and processes 4 code tiles (64 codes) per
// pass with 4 accumulators: one ds_load_b128 pair of A feeds 4 WMMAs, so the
// compiler cannot profitably hoist A across the n0 loop (no scratch spill).
// score = ||E_k||^2 - 2 * (r . E_k)   (||r||^2 is row-constant -> dropped)
// ---------------------------------------------------------------------------
__global__ void rvq_argmin_kernel(const u16* __restrict__ Rbf, const u16* __restrict__ EbfT,
                                  const float* __restrict__ norms, int* __restrict__ idxArr,
                                  float* __restrict__ outIds, int level) {
    extern __shared__ char smem[];
    u16* As = (u16*)smem;                          // [64][768] bf16 residual tile

    const int lane = threadIdx.x & 31;
    const int wave = threadIdx.x >> 5;             // 0..3
    const int h    = lane >> 4;                    // lane half
    const int ln   = lane & 15;                    // m (A) / n (B,C) index

    // stage 64 contiguous residual rows into LDS (vectorized b128 copies)
    {
        const uint4* src = (const uint4*)(Rbf + (size_t)blockIdx.x * 64 * DD);
        uint4* dst = (uint4*)smem;
        const int nvec = 64 * DD / 8;
        for (int i = threadIdx.x; i < nvec; i += blockDim.x) dst[i] = src[i];
    }
    __syncthreads();

    const int rowBase = wave * 16;
    // A fragment (this code's contiguous k order, same as B): lane m=ln; VGPRs hold K=16h+{0..15} contiguous
    const u16* aptr = As + (rowBase + ln) * DD + 16 * h;

    float minv[8];
    int   mini[8];
#pragma unroll
    for (int r = 0; r < 8; ++r) { minv[r] = 3.4e38f; mini[r] = 0x7fffffff; }

    for (int n0 = 0; n0 < KK; n0 += 16 * NT) {
        // B fragment (same contiguous k order as A): lane n=ln; VGPRs hold K=16h+{0..15} contiguous.
        // The NT tiles sit at constant strides -> fold into instruction offsets.
        const u16* bbase = EbfT + (size_t)(n0 + ln) * DD + 16 * h;
        __builtin_prefetch(bbase + 16 * NT * DD, 0, 3);   // next pass's codes

        v8f c[NT];
#pragma unroll
        for (int t = 0; t < NT; ++t) c[t] = (v8f){};

#pragma unroll 2
        for (int kk = 0; kk < DD; kk += 32) {
            Frag a;
            a.q[0] = *(const uint4*)(aptr + kk);          // ds_load_b128
            a.q[1] = *(const uint4*)(aptr + kk + 8);
#pragma unroll
            for (int t = 0; t < NT; ++t) {
                Frag b;
                b.q[0] = *(const uint4*)(bbase + t * 16 * DD + kk);      // global_load_b128
                b.q[1] = *(const uint4*)(bbase + t * 16 * DD + kk + 8);
                c[t] = __builtin_amdgcn_wmma_f32_16x16x32_bf16(
                           false, a.bf, false, b.bf, (short)0, c[t], false, false);
            }
        }

#pragma unroll
        for (int t = 0; t < NT; ++t) {
            const float nrm = norms[n0 + 16 * t + ln];
            const int   cid = n0 + 16 * t + ln;
#pragma unroll
            for (int r = 0; r < 8; ++r) {
                float sc = nrm - 2.0f * c[t][r];          // C/D: lane -> (M = r + 8h, N = ln)
                if (sc < minv[r] || (sc == minv[r] && cid < mini[r])) {
                    minv[r] = sc; mini[r] = cid;
                }
            }
        }
    }

    // reduce argmin across the 16 lanes sharing each row group
#pragma unroll
    for (int off = 8; off > 0; off >>= 1) {
#pragma unroll
        for (int r = 0; r < 8; ++r) {
            float ov = __shfl_xor(minv[r], off, 32);
            int   oi = __shfl_xor(mini[r], off, 32);
            if (ov < minv[r] || (ov == minv[r] && oi < mini[r])) {
                minv[r] = ov; mini[r] = oi;
            }
        }
    }

    if (ln == 0) {                                        // lanes 0 and 16 write
#pragma unroll
        for (int r = 0; r < 8; ++r) {
            int row = blockIdx.x * 64 + rowBase + h * 8 + r;   // C layout: M = r + 8h
            idxArr[row] = mini[r];
            outIds[(size_t)row * LL + level] = (float)mini[r];
        }
    }
}

// ---------------------------------------------------------------------------
// Per level: gather f32 codeword from row-major ET32 (coalesced: each sample
// reads a contiguous 3KB row), MSE loss, residual update, quantized_x +=.
// ---------------------------------------------------------------------------
__global__ void rvq_update_kernel(const float* __restrict__ ET32, const int* __restrict__ idxArr,
                                  float* __restrict__ Rf32, float* __restrict__ qx,
                                  float* __restrict__ loss) {
    int t = blockIdx.x * blockDim.x + threadIdx.x;
    float dsq = 0.0f;
    if (t < BB * DD) {
        int b = t / DD, d = t % DD;
        int ind = idxArr[b];
        float q = ET32[(size_t)ind * DD + d];
        float r = Rf32[t];
        float df = q - r;
        dsq = df * df;
        Rf32[t] = r - q;        // new residual
        qx[t]  += q;            // accumulate quantized_x
    }
    for (int off = 16; off > 0; off >>= 1) dsq += __shfl_down(dsq, off, 32);
    __shared__ float wsum[8];
    int lane = threadIdx.x & 31, wid = threadIdx.x >> 5;
    if (lane == 0) wsum[wid] = dsq;
    __syncthreads();
    if (threadIdx.x == 0) {
        float s = 0.0f;
        for (int w = 0; w < (int)(blockDim.x >> 5); ++w) s += wsum[w];
        atomicAdd(loss, s * (1.0f / ((float)BB * (float)DD)));
    }
}

// ---------------------------------------------------------------------------
extern "C" void kernel_launch(void* const* d_in, const int* in_sizes, int n_in,
                              void* d_out, int out_size, void* d_ws, size_t ws_size,
                              hipStream_t stream) {
    const float* x      = (const float*)d_in[0];   // [B, D]
    const float* embeds = (const float*)d_in[1];   // [L, D, K]
    const float* cs     = (const float*)d_in[2];   // [L, K]
    float* out = (float*)d_out;

    char* ws = (char*)d_ws;
    size_t off = 0;
    float* Rf32 = (float*)(ws + off); off += (size_t)BB * DD * 4;   // 25 MB
    u16*   Rbf  = (u16*)  (ws + off); off += (size_t)BB * DD * 2;   // 12 MB
    u16*   EbfT = (u16*)  (ws + off); off += (size_t)KK * DD * 2;   // 12 MB
    float* ET32 = (float*)(ws + off); off += (size_t)KK * DD * 4;   // 25 MB
    float* nrm  = (float*)(ws + off); off += (size_t)KK * 4;
    int*   idxA = (int*)  (ws + off); off += (size_t)BB * 4;

    float* outIds = out;                            // [B, L]
    float* qx     = out + (size_t)BB * LL;          // [B, D]
    float* smalls = out + (size_t)BB * LL + (size_t)BB * DD;
    float* loss   = smalls + 1;

    const int BD = BB * DD;
    rvq_init_kernel<<<(BD + 255) / 256, 256, 0, stream>>>(x, Rf32, qx, smalls, loss);

    for (int l = 0; l < LL; ++l) {
        const float* E   = embeds + (size_t)l * DD * KK;
        const float* csl = cs + (size_t)l * KK;

        rvq_transpose_kernel<<<dim3(KK / 32, DD / 32), dim3(32, 8), 0, stream>>>(E, EbfT, ET32);
        rvq_norms_kernel<<<(KK + 255) / 256, 256, 0, stream>>>(E, csl, nrm, smalls);
        rvq_tobf16_kernel<<<(BD + 255) / 256, 256, 0, stream>>>(Rf32, Rbf);
        rvq_argmin_kernel<<<BB / 64, 128, 64 * DD * sizeof(u16), stream>>>(
            Rbf, EbfT, nrm, idxA, outIds, l);
        rvq_update_kernel<<<(BD + 255) / 256, 256, 0, stream>>>(ET32, idxA, Rf32, qx, loss);
    }
}
